// PGAttentionLeVit_18957985644959
// MI455X (gfx1250) — matrix operand order repaired and checked by reading, not compiled, not run on hardware
//
#include <hip/hip_runtime.h>
#include <hip/hip_bf16.h>

typedef __attribute__((ext_vector_type(16))) _Float16 v16h;
typedef __attribute__((ext_vector_type(8)))  _Float16 h8;
typedef __attribute__((ext_vector_type(8)))  float    v8f;

#define WMMA_F32_16x16x32_F16(A, B, C) \
  __builtin_amdgcn_wmma_f32_16x16x32_f16(false, (A), false, (B), (short)0, (C), false, false)

// ---------------------------------------------------------------------------
// Problem constants
// ---------------------------------------------------------------------------
static constexpr int BATCH   = 8;
static constexpr int NHEADS  = 12;
static constexpr int KD      = 64;     // key dim
static constexpr int DV      = 256;    // value dim (ATTN_RATIO*KD)
static constexpr int NTOK    = 1024;   // RES*RES
static constexpr int DIM     = 768;
static constexpr int QKV_OUT = NHEADS * (2 * KD + DV);  // 4608
static constexpr int HSTRIDE = 2 * KD + DV;             // 384
static constexpr int DH      = NHEADS * DV;             // 3072
static constexpr int MROWS   = BATCH * NTOK;            // 8192

__device__ __forceinline__ float quant_f(float x, float s, float inv) {
  return rintf(x * s) * inv;  // round-to-nearest-even, matches jnp.round
}

// ---------------------------------------------------------------------------
// Fragment loaders (CDNA5 wave32 WMMA layouts, 16-bit operands, K=32)
// A 16x32: lane L holds row M=L%16; K = koffA + j + (j>=8?8:0), koffA=(L&16)?8:0
//          -> two contiguous 8-half (16B) runs per lane
// B 32x16: lane L holds col N=L%16; K = koffB + j + (j>=8?8:0), koffB=(L&16)?8:0
//          -> two contiguous 8-half (16B) runs per lane
// Both loaders emit 2x b128 loads.
// ---------------------------------------------------------------------------
__device__ __forceinline__ v16h fragA_h(const _Float16* __restrict__ M, int ld,
                                        int m0, int k0) {
  int lane = threadIdx.x & 31;
  const _Float16* p = M + (long)(m0 + (lane & 15)) * ld + k0 + ((lane & 16) ? 8 : 0);
  h8 lo = *(const h8*)p;
  h8 hi = *(const h8*)(p + 16);
  return __builtin_shufflevector(lo, hi, 0, 1, 2, 3, 4, 5, 6, 7,
                                 8, 9, 10, 11, 12, 13, 14, 15);
}

// B = W^T from row-major W[n][k] (lane's col n fixed, K contiguous)
__device__ __forceinline__ v16h fragB_hT(const _Float16* __restrict__ W, int ld,
                                         int n0, int k0) {
  int lane = threadIdx.x & 31;
  const _Float16* p = W + (long)(n0 + (lane & 15)) * ld + k0 + ((lane & 16) ? 8 : 0);
  h8 lo = *(const h8*)p;
  h8 hi = *(const h8*)(p + 16);
  return __builtin_shufflevector(lo, hi, 0, 1, 2, 3, 4, 5, 6, 7,
                                 8, 9, 10, 11, 12, 13, 14, 15);
}

// elementwise 4-bit fake-quantize of a fragment (PGABITS = 4 -> scale 8)
__device__ __forceinline__ v16h quant4_frag(v16h f) {
  v16h r;
#pragma unroll
  for (int j = 0; j < 16; ++j)
    r[j] = (_Float16)(rintf((float)f[j] * 8.0f) * 0.125f);
  return r;
}

// ---------------------------------------------------------------------------
// Kernel 0: elementwise 8-bit fake-quantize f32 -> fp16 (pre-pass)
// ---------------------------------------------------------------------------
__global__ __launch_bounds__(256) void quant8_kernel(
    const float* __restrict__ src, _Float16* __restrict__ dst, long n) {
  long i = ((long)blockIdx.x * blockDim.x + threadIdx.x) * 4;
  if (i >= n) return;
  float4 v = *(const float4*)(src + i);
  dst[i + 0] = (_Float16)quant_f(v.x, 128.0f, 1.0f / 128.0f);
  dst[i + 1] = (_Float16)quant_f(v.y, 128.0f, 1.0f / 128.0f);
  dst[i + 2] = (_Float16)quant_f(v.z, 128.0f, 1.0f / 128.0f);
  dst[i + 3] = (_Float16)quant_f(v.w, 128.0f, 1.0f / 128.0f);
}

// ---------------------------------------------------------------------------
// Kernel 1: QKV GEMM  h = xq @ wq^T, h = q8(h*g + b) -> split Q/K/V^T fp16
// block: 256 thr (8 waves, 2x4), wave does 32x32 (2x2 WMMA tiles)
// block tile 64 rows x 128 cols; grid (8192/64, 4608/128)
// ---------------------------------------------------------------------------
__global__ __launch_bounds__(256) void qkv_kernel(
    const _Float16* __restrict__ xq, const _Float16* __restrict__ wq,
    const float* __restrict__ g, const float* __restrict__ bb,
    _Float16* __restrict__ Q, _Float16* __restrict__ Kb, _Float16* __restrict__ Vt) {
  int lane = threadIdx.x & 31, wave = threadIdx.x >> 5;
  int m0 = blockIdx.x * 64 + (wave >> 2) * 32;
  int n0 = blockIdx.y * 128 + (wave & 3) * 32;

  v8f acc00 = {}, acc01 = {}, acc10 = {}, acc11 = {};
  for (int k0 = 0; k0 < DIM; k0 += 32) {
    v16h a0 = fragA_h(xq, DIM, m0, k0);
    v16h a1 = fragA_h(xq, DIM, m0 + 16, k0);
    v16h b0 = fragB_hT(wq, DIM, n0, k0);
    v16h b1 = fragB_hT(wq, DIM, n0 + 16, k0);
    acc00 = WMMA_F32_16x16x32_F16(a0, b0, acc00);
    acc01 = WMMA_F32_16x16x32_F16(a0, b1, acc01);
    acc10 = WMMA_F32_16x16x32_F16(a1, b0, acc10);
    acc11 = WMMA_F32_16x16x32_F16(a1, b1, acc11);
  }

  int hi = (lane & 16) ? 8 : 0, l15 = lane & 15;
  v8f* accs[4] = {&acc00, &acc01, &acc10, &acc11};
#pragma unroll
  for (int i = 0; i < 2; ++i) {
#pragma unroll
    for (int j = 0; j < 2; ++j) {
      v8f& acc = *accs[i * 2 + j];
      int col = n0 + j * 16 + l15;
      float gg = g[col], bbv = bb[col];
      int head = col / HSTRIDE;
      int rr = col - head * HSTRIDE;
#pragma unroll
      for (int r = 0; r < 8; ++r) {
        int row = m0 + i * 16 + r + hi;
        float v = quant_f(acc[r] * gg + bbv, 128.0f, 1.0f / 128.0f);
        int b_ = row >> 10, n = row & 1023;
        long bh = (long)(b_ * NHEADS + head);
        _Float16 hv = (_Float16)v;
        if (rr < KD)           Q[(bh * NTOK + n) * KD + rr] = hv;
        else if (rr < 2 * KD)  Kb[(bh * NTOK + n) * KD + (rr - KD)] = hv;
        else                   Vt[(bh * DV + (rr - 2 * KD)) * NTOK + n] = hv;  // V^T
      }
    }
  }
}

// ---------------------------------------------------------------------------
// Kernel 2: masked two-softmax attention + attn@V + hardswish + q8
// block: 128 thr (4 waves), one (b,h) x 16 query rows; grid (96, 64)
// ---------------------------------------------------------------------------
__global__ __launch_bounds__(128) void attn_kernel(
    const _Float16* __restrict__ Q, const _Float16* __restrict__ Km,
    const _Float16* __restrict__ Vt, const float* __restrict__ attn_biases,
    const int* __restrict__ bias_idxs, _Float16* __restrict__ Oall, int n_uniq) {
  __shared__ __align__(16) _Float16 att_s[16 * NTOK];   // quantized probs (32 KB)
  __shared__ float red_m[4][16], red_s[4][16];
  __shared__ float row_m1[16], row_s1[16], row_m2[16], row_s2[16];

  int lane = threadIdx.x & 31, wave = threadIdx.x >> 5;
  int l15 = lane & 15, hi = (lane & 16) ? 8 : 0;
  int bh = blockIdx.x;
  int b_ = bh / NHEADS, h = bh - b_ * NHEADS;
  int m0 = blockIdx.y * 16;

  const _Float16* Qp = Q + (long)bh * NTOK * KD;
  const _Float16* Kp = Km + (long)bh * NTOK * KD;
  const _Float16* Vp = Vt + (long)bh * DV * NTOK;       // V^T: [d][m]
  const float* biasRow = attn_biases + (long)h * n_uniq;
  const float scale = 0.125f;  // KD^-0.5

  // q fragments (held across all phases) + 4-bit msb versions
  v16h aq0 = fragA_h(Qp, KD, m0, 0), aq1 = fragA_h(Qp, KD, m0, 32);
  v16h am0 = quant4_frag(aq0),       am1 = quant4_frag(aq1);

  float mx[8], sm[8];

  // ---- Phase 1: online softmax stats of msb logits -------------------------
#pragma unroll
  for (int r = 0; r < 8; ++r) { mx[r] = -1e30f; sm[r] = 0.0f; }
  for (int ct = wave; ct < 64; ct += 4) {
    int col0 = ct * 16;
    v16h bm0 = quant4_frag(fragB_hT(Kp, KD, col0, 0));
    v16h bm1 = quant4_frag(fragB_hT(Kp, KD, col0, 32));
    v8f c = {};
    c = WMMA_F32_16x16x32_F16(am0, bm0, c);
    c = WMMA_F32_16x16x32_F16(am1, bm1, c);
    int col = col0 + l15;
#pragma unroll
    for (int r = 0; r < 8; ++r) {
      int row = m0 + r + hi;
      float bias = biasRow[bias_idxs[(long)row * NTOK + col]];
      float s = c[r] * scale + bias;
      float nm = fmaxf(mx[r], s);
      sm[r] = sm[r] * __expf(mx[r] - nm) + __expf(s - nm);
      mx[r] = nm;
    }
  }
#pragma unroll
  for (int d = 1; d < 16; d <<= 1)
#pragma unroll
    for (int r = 0; r < 8; ++r) {
      float om = __shfl_xor(mx[r], d), os = __shfl_xor(sm[r], d);
      float nm = fmaxf(mx[r], om);
      sm[r] = sm[r] * __expf(mx[r] - nm) + os * __expf(om - nm);
      mx[r] = nm;
    }
  if (l15 == 0)
#pragma unroll
    for (int r = 0; r < 8; ++r) { red_m[wave][hi + r] = mx[r]; red_s[wave][hi + r] = sm[r]; }
  __syncthreads();
  if (threadIdx.x < 16) {
    float m = -1e30f, s = 0.0f;
    for (int w = 0; w < 4; ++w) m = fmaxf(m, red_m[w][threadIdx.x]);
    for (int w = 0; w < 4; ++w) s += red_s[w][threadIdx.x] * __expf(red_m[w][threadIdx.x] - m);
    row_m1[threadIdx.x] = m; row_s1[threadIdx.x] = s;
  }
  __syncthreads();

  // ---- Phase 2: stats of masked full logits --------------------------------
#pragma unroll
  for (int r = 0; r < 8; ++r) { mx[r] = -1e30f; sm[r] = 0.0f; }
  for (int ct = wave; ct < 64; ct += 4) {
    int col0 = ct * 16;
    v16h bk0 = fragB_hT(Kp, KD, col0, 0);
    v16h bk1 = fragB_hT(Kp, KD, col0, 32);
    v16h bm0 = quant4_frag(bk0), bm1 = quant4_frag(bk1);
    v8f cm = {}, cf = {};
    cm = WMMA_F32_16x16x32_F16(am0, bm0, cm);
    cm = WMMA_F32_16x16x32_F16(am1, bm1, cm);
    cf = WMMA_F32_16x16x32_F16(aq0, bk0, cf);
    cf = WMMA_F32_16x16x32_F16(aq1, bk1, cf);
    int col = col0 + l15;
#pragma unroll
    for (int r = 0; r < 8; ++r) {
      int row = m0 + r + hi;
      float bias = biasRow[bias_idxs[(long)row * NTOK + col]];
      float smsb = cm[r] * scale + bias;
      float p1 = __expf(smsb - row_m1[r + hi]) / row_s1[r + hi];
      float logit = (p1 > 0.99f) ? (cf[r] * scale + bias) : 0.0f;
      float nm = fmaxf(mx[r], logit);
      sm[r] = sm[r] * __expf(mx[r] - nm) + __expf(logit - nm);
      mx[r] = nm;
    }
  }
#pragma unroll
  for (int d = 1; d < 16; d <<= 1)
#pragma unroll
    for (int r = 0; r < 8; ++r) {
      float om = __shfl_xor(mx[r], d), os = __shfl_xor(sm[r], d);
      float nm = fmaxf(mx[r], om);
      sm[r] = sm[r] * __expf(mx[r] - nm) + os * __expf(om - nm);
      mx[r] = nm;
    }
  if (l15 == 0)
#pragma unroll
    for (int r = 0; r < 8; ++r) { red_m[wave][hi + r] = mx[r]; red_s[wave][hi + r] = sm[r]; }
  __syncthreads();
  if (threadIdx.x < 16) {
    float m = -1e30f, s = 0.0f;
    for (int w = 0; w < 4; ++w) m = fmaxf(m, red_m[w][threadIdx.x]);
    for (int w = 0; w < 4; ++w) s += red_s[w][threadIdx.x] * __expf(red_m[w][threadIdx.x] - m);
    row_m2[threadIdx.x] = m; row_s2[threadIdx.x] = s;
  }
  __syncthreads();

  // ---- Phase 3a: materialize q8(softmax) probs into LDS --------------------
  for (int ct = wave; ct < 64; ct += 4) {
    int col0 = ct * 16;
    v16h bk0 = fragB_hT(Kp, KD, col0, 0);
    v16h bk1 = fragB_hT(Kp, KD, col0, 32);
    v16h bm0 = quant4_frag(bk0), bm1 = quant4_frag(bk1);
    v8f cm = {}, cf = {};
    cm = WMMA_F32_16x16x32_F16(am0, bm0, cm);
    cm = WMMA_F32_16x16x32_F16(am1, bm1, cm);
    cf = WMMA_F32_16x16x32_F16(aq0, bk0, cf);
    cf = WMMA_F32_16x16x32_F16(aq1, bk1, cf);
    int col = col0 + l15;
#pragma unroll
    for (int r = 0; r < 8; ++r) {
      int row = m0 + r + hi;
      float bias = biasRow[bias_idxs[(long)row * NTOK + col]];
      float smsb = cm[r] * scale + bias;
      float p1 = __expf(smsb - row_m1[r + hi]) / row_s1[r + hi];
      float logit = (p1 > 0.99f) ? (cf[r] * scale + bias) : 0.0f;
      float p = __expf(logit - row_m2[r + hi]) / row_s2[r + hi];
      att_s[(r + hi) * NTOK + col] = (_Float16)quant_f(p, 128.0f, 1.0f / 128.0f);
    }
  }
  __syncthreads();

  // ---- Phase 3b: O = attn @ V (each wave owns 64 of 256 out cols) ----------
  int n0w = wave * 64;
  v8f o[4] = {};
  for (int mt = 0; mt < 32; ++mt) {
    v16h a = fragA_h(att_s, NTOK, 0, mt * 32);   // ds_load_b128 from LDS
#pragma unroll
    for (int t = 0; t < 4; ++t) {
      v16h bv = fragB_hT(Vp, NTOK, n0w + t * 16, mt * 32);  // contiguous V^T rows
      o[t] = WMMA_F32_16x16x32_F16(a, bv, o[t]);
    }
  }
#pragma unroll
  for (int t = 0; t < 4; ++t)
#pragma unroll
    for (int r = 0; r < 8; ++r) {
      int row = m0 + r + hi;
      int col = n0w + t * 16 + l15;
      float v = o[t][r];
      float hsw = v * fminf(fmaxf(v + 3.0f, 0.0f), 6.0f) * (1.0f / 6.0f);
      Oall[((long)b_ * NTOK + row) * DH + h * DV + col] =
          (_Float16)quant_f(hsw, 128.0f, 1.0f / 128.0f);
    }
}

// ---------------------------------------------------------------------------
// Kernel 3: proj GEMM  out = Oall @ pwq^T * g + b   (f32 output)
// block: 8 waves, wave 32x32; block 64x128; grid (8192/64, 768/128)
// ---------------------------------------------------------------------------
__global__ __launch_bounds__(256) void proj_kernel(
    const _Float16* __restrict__ A, const _Float16* __restrict__ pwq,
    const float* __restrict__ g, const float* __restrict__ bb,
    float* __restrict__ out) {
  int lane = threadIdx.x & 31, wave = threadIdx.x >> 5;
  int m0 = blockIdx.x * 64 + (wave >> 2) * 32;
  int n0 = blockIdx.y * 128 + (wave & 3) * 32;

  v8f acc00 = {}, acc01 = {}, acc10 = {}, acc11 = {};
  for (int k0 = 0; k0 < DH; k0 += 32) {
    v16h a0 = fragA_h(A, DH, m0, k0);
    v16h a1 = fragA_h(A, DH, m0 + 16, k0);
    v16h b0 = fragB_hT(pwq, DH, n0, k0);
    v16h b1 = fragB_hT(pwq, DH, n0 + 16, k0);
    acc00 = WMMA_F32_16x16x32_F16(a0, b0, acc00);
    acc01 = WMMA_F32_16x16x32_F16(a0, b1, acc01);
    acc10 = WMMA_F32_16x16x32_F16(a1, b0, acc10);
    acc11 = WMMA_F32_16x16x32_F16(a1, b1, acc11);
  }
  int hi = (lane & 16) ? 8 : 0, l15 = lane & 15;
  v8f* accs[4] = {&acc00, &acc01, &acc10, &acc11};
#pragma unroll
  for (int i = 0; i < 2; ++i)
#pragma unroll
    for (int j = 0; j < 2; ++j) {
      v8f& acc = *accs[i * 2 + j];
      int col = n0 + j * 16 + l15;
      float gg = g[col], bbv = bb[col];
#pragma unroll
      for (int r = 0; r < 8; ++r) {
        int row = m0 + i * 16 + r + hi;
        out[(long)row * DIM + col] = acc[r] * gg + bbv;
      }
    }
}

// ---------------------------------------------------------------------------
extern "C" void kernel_launch(void* const* d_in, const int* in_sizes, int n_in,
                              void* d_out, int out_size, void* d_ws, size_t ws_size,
                              hipStream_t stream) {
  const float* x           = (const float*)d_in[0];
  const float* qkv_w       = (const float*)d_in[1];
  const float* qkv_g       = (const float*)d_in[2];
  const float* qkv_b       = (const float*)d_in[3];
  const float* proj_w      = (const float*)d_in[4];
  const float* proj_g      = (const float*)d_in[5];
  const float* proj_b      = (const float*)d_in[6];
  const float* attn_biases = (const float*)d_in[7];
  const int*   bias_idxs   = (const int*)d_in[8];
  float* out = (float*)d_out;

  int n_uniq = in_sizes[7] / NHEADS;

  // workspace layout (fp16 halves)
  _Float16* p = (_Float16*)d_ws;
  long xq_n  = (long)MROWS * DIM;        _Float16* xq  = p; p += xq_n;
  long wq_n  = (long)QKV_OUT * DIM;      _Float16* wq  = p; p += wq_n;
  long pwq_n = (long)DIM * DH;           _Float16* pwq = p; p += pwq_n;
  long qk_n  = (long)BATCH * NHEADS * NTOK * KD;
  _Float16* Q  = p; p += qk_n;
  _Float16* K  = p; p += qk_n;
  long v_n   = (long)BATCH * NHEADS * NTOK * DV;
  _Float16* Vt = p; p += v_n;
  _Float16* O  = p;                      // 8*1024*3072 halves

  // pre-quantize f32 -> fp16 (8-bit fake quant grid)
  quant8_kernel<<<dim3((unsigned)((xq_n / 4 + 255) / 256)), 256, 0, stream>>>(x, xq, xq_n);
  quant8_kernel<<<dim3((unsigned)((wq_n / 4 + 255) / 256)), 256, 0, stream>>>(qkv_w, wq, wq_n);
  quant8_kernel<<<dim3((unsigned)((pwq_n / 4 + 255) / 256)), 256, 0, stream>>>(proj_w, pwq, pwq_n);

  qkv_kernel<<<dim3(MROWS / 64, QKV_OUT / 128), 256, 0, stream>>>(
      xq, wq, qkv_g, qkv_b, Q, K, Vt);

  attn_kernel<<<dim3(BATCH * NHEADS, NTOK / 16), 128, 0, stream>>>(
      Q, K, Vt, attn_biases, bias_idxs, O, n_uniq);

  proj_kernel<<<dim3(MROWS / 64, DIM / 128), 256, 0, stream>>>(
      O, pwq, proj_g, proj_b, out);

  (void)n_in; (void)out_size; (void)ws_size;
}
